// MLP_Param_7249904795891
// MI455X (gfx1250) — matrix operand order repaired and checked by reading, not compiled, not run on hardware
//
#include <hip/hip_runtime.h>

// ---------------------------------------------------------------------------
// Dynamic-weight MLP:  out = einsum('ni,nd,dio->no', x, q, W1) + q @ b1
// Rewritten as a single GEMM:  C[4096,256] = A[4096,16448] @ B[16448,256]
//   A[n, d*256+i] = q[n,d]*x[n,i]   (k <  16384)
//   A[n, 16384+d] = q[n,d]          (k >= 16384, bias fold)
//   B rows 0..16383 = W1 flattened [d*256+i, o];  rows 16384.. = b1[d, o]
// bf16 WMMA (v_wmma_f32_16x16x32_bf16), fp32 accumulate.
// Wave tile 32x64 (2x4 WMMA); B addressed via one base pointer + immediate
// offsets (24-bit IOFFSET) so address VALU ~1 add per 8 k-tiles.
// ---------------------------------------------------------------------------

typedef __attribute__((ext_vector_type(16))) __bf16 v16bf;
typedef __attribute__((ext_vector_type(8)))  float  v8f;
typedef __attribute__((ext_vector_type(4)))  int    i4;
typedef __attribute__((ext_vector_type(2)))  int    i2;
typedef __attribute__((ext_vector_type(4)))  float  f4;

#define N_SAMP 4096
#define D_IN   256
#define D_OUT  256
#define Q_DIM  64
#define K_TOT  (Q_DIM * D_IN + Q_DIM)   // 16448
#define KTILES (K_TOT / 32)             // 514 k-tiles of 32
#define KT_BYTES 16384                  // bytes per k-tile row of packed B (16 tiles * 1 KB)

// float -> bf16 raw bits, round-to-nearest-even (NaN not special-cased)
static __device__ __forceinline__ unsigned f2bf_raw(float f) {
  unsigned u = __builtin_bit_cast(unsigned, f);
  return (u + 0x7FFFu + ((u >> 16) & 1u)) >> 16;
}
static __device__ __forceinline__ int pack2bf(float lo, float hi) {
  return (int)(f2bf_raw(lo) | (f2bf_raw(hi) << 16));
}
// packed bf16 x bf16 multiply (VOP3P) — one VALU op scales a dword of A
static __device__ __forceinline__ int pk_mul_bf16(int a, int s) {
  int d;
  asm("v_pk_mul_bf16 %0, %1, %2" : "=v"(d) : "v"(a), "v"(s));
  return d;
}

union AFrag {           // 16 bf16 = one WMMA A/B operand (8 VGPRs)
  i4    h[2];
  int   i[8];
  v16bf v;
};

static __device__ __forceinline__ v8f wmma_bf16(const AFrag& a, const AFrag& b, v8f c) {
  return __builtin_amdgcn_wmma_f32_16x16x32_bf16(false, a.v, false, b.v,
                                                 (short)0, c, false, false);
}

// ---------------------------------------------------------------------------
// Prepass: pack W1 (+ b1 as extra K rows) into fragment-native bf16 layout.
// B-tile (32x16 bf16) for k-tile kt, n-tile nt: lane's 8 dwords stored
// contiguously at ((kt*16+nt)*32 + lane)*32 bytes, following the ISA 16-bit
// B layout: lane<16 -> col=lane, K = kt*32 + {0..7,16..23}; lane>=16 -> +8 K.
// ---------------------------------------------------------------------------
__global__ __launch_bounds__(256) void pack_w(const float* __restrict__ W,
                                              const float* __restrict__ b,
                                              unsigned short* __restrict__ wp) {
  int gid = blockIdx.x * 256 + threadIdx.x;          // one per (kt, nt, lane)
  if (gid >= KTILES * 16 * 32) return;
  int lane = gid & 31;
  int nt   = (gid >> 5) & 15;
  int kt   = gid >> 9;
  int c     = nt * 16 + (lane & 15);
  int kbase = kt * 32 + ((lane & 16) ? 8 : 0);
  int regs[8];
#pragma unroll
  for (int j = 0; j < 8; ++j) {
    int k0 = kbase + 2 * j + (j >= 4 ? 8 : 0);
    int k1 = k0 + 1;
    float f0 = (k0 < Q_DIM * D_IN) ? W[(size_t)k0 * D_OUT + c]
                                   : b[(size_t)(k0 - Q_DIM * D_IN) * D_OUT + c];
    float f1 = (k1 < Q_DIM * D_IN) ? W[(size_t)k1 * D_OUT + c]
                                   : b[(size_t)(k1 - Q_DIM * D_IN) * D_OUT + c];
    regs[j] = pack2bf(f0, f1);
  }
  i4* dst = (i4*)(wp + (size_t)gid * 16);
  dst[0] = (i4){regs[0], regs[1], regs[2], regs[3]};
  dst[1] = (i4){regs[4], regs[5], regs[6], regs[7]};
}

// ---------------------------------------------------------------------------
// Main GEMM. Block tile 64(M) x 128(N), 4 waves (128 thr), wave tile 32x64
// (2x4 WMMA tiles -> 8 WMMA per k-step against 2 A-frag builds).
// x-block staged in LDS as bf16 (reused across all 64 d slices); A fragments
// built from LDS (immediate ds offsets) and scaled with v_pk_mul_bf16;
// B fragments loaded from a single advancing base pointer with immediate
// global offsets. Bias = 2 extra k-tiles continuing the same pointer.
// ---------------------------------------------------------------------------
__global__ __launch_bounds__(128) void dynw_gemm(const float* __restrict__ x,
                                                 const float* __restrict__ q,
                                                 const unsigned short* __restrict__ wp,
                                                 float* __restrict__ out) {
  __shared__ alignas(16) unsigned short xs[64 * 256];  // x block, bf16 (32 KB)
  __shared__ alignas(16) float          qf[64 * 64];   // q block, f32 (16 KB)
  __shared__ alignas(16) unsigned short qb[64 * 64];   // q block, bf16 (8 KB)

  const int tid   = threadIdx.x;
  const int lane  = tid & 31;
  const int wid   = tid >> 5;      // 0..3
  const int waveM = wid & 1;       // 2 waves along M (32 rows each)
  const int waveN = wid >> 1;      // 2 waves along N (64 cols each)
  const int m0 = blockIdx.y * 64;
  const int n0 = blockIdx.x * 128;

  // ---- stage x (64x256 f32 -> bf16) and q (64x64, f32 + bf16) into LDS ----
  const f4* x4 = (const f4*)(x + (size_t)m0 * D_IN);
#pragma unroll
  for (int it = 0; it < 32; ++it) {
    int fi = it * 128 + tid;               // float4 index within 64x256 block
    f4 v = x4[fi];
    int row = fi >> 6, c4 = fi & 63;
    i2 p;
    p.x = pack2bf(v.x, v.y);
    p.y = pack2bf(v.z, v.w);
    *(i2*)&xs[row * 256 + c4 * 4] = p;
  }
  const f4* q4 = (const f4*)(q + (size_t)m0 * Q_DIM);
#pragma unroll
  for (int it = 0; it < 8; ++it) {
    int fi = it * 128 + tid;               // float4 index within 64x64 block
    f4 v = q4[fi];
    int row = fi >> 4, c4 = fi & 15;
    *(f4*)&qf[row * 64 + c4 * 4] = v;
    i2 p;
    p.x = pack2bf(v.x, v.y);
    p.y = pack2bf(v.z, v.w);
    *(i2*)&qb[row * 64 + c4 * 4] = p;
  }
  __syncthreads();

  // ISA 16-bit A layout: lane<16 -> K {0..7,16..23}; lane>=16 -> +8
  const int k8  = (lane & 16) ? 8 : 0;
  const int rA0 = waveM * 32 + (lane & 15);   // row of A tile 0 (local)
  const int rA1 = rA0 + 16;                   // row of A tile 1
  const int nt0 = blockIdx.x * 8 + waveN * 4; // first global n-tile of 4

  // Fixed per-wave pointers; all subsequent addressing via immediate offsets.
  const unsigned short* xa0 = xs + rA0 * 256 + k8;
  const unsigned short* xa1 = xs + rA1 * 256 + k8;
  const float*          qs0 = qf + rA0 * 64;
  const float*          qs1 = qf + rA1 * 64;
  // Base of this wave's B fragments at k-tile 0 (lane's 32-byte slice).
  const char* bptr = (const char*)wp + ((size_t)nt0 * 32 + lane) * 32;

  v8f acc[2][4] = {};

  for (int d = 0; d < Q_DIM; ++d) {
    const int s0p = (int)(f2bf_raw(qs0[d]) * 0x10001u);  // splat bf16 scale
    const int s1p = (int)(f2bf_raw(qs1[d]) * 0x10001u);
#pragma unroll
    for (int ib = 0; ib < 8; ++ib) {
      AFrag a0, a1;
      a0.h[0] = *(const i4*)(xa0 + ib * 32);
      a0.h[1] = *(const i4*)(xa0 + ib * 32 + 16);
      a1.h[0] = *(const i4*)(xa1 + ib * 32);
      a1.h[1] = *(const i4*)(xa1 + ib * 32 + 16);
      AFrag bf[4];
#pragma unroll
      for (int u = 0; u < 4; ++u) {   // immediate offsets: <= 118 KB, fits IOFFSET
        bf[u].h[0] = *(const i4*)(bptr + ib * KT_BYTES + u * 1024);
        bf[u].h[1] = *(const i4*)(bptr + ib * KT_BYTES + u * 1024 + 16);
      }
#pragma unroll
      for (int j = 0; j < 8; ++j) {
        a0.i[j] = pk_mul_bf16(a0.i[j], s0p);   // A = q[n,d] * x[n,i]
        a1.i[j] = pk_mul_bf16(a1.i[j], s1p);
      }
#pragma unroll
      for (int u = 0; u < 4; ++u) {
        acc[0][u] = wmma_bf16(a0, bf[u], acc[0][u]);
        acc[1][u] = wmma_bf16(a1, bf[u], acc[1][u]);
      }
    }
    bptr += 8 * KT_BYTES;   // advance one d-slice (8 k-tiles): 1 add per 64 WMMAs
  }

  // ---- bias fold: 2 extra k-tiles where A = q (bf16), B = packed b1 ----
  // bptr now points at k-tile 512.
  const unsigned short* qba0 = qb + rA0 * 64 + k8;
  const unsigned short* qba1 = qb + rA1 * 64 + k8;
#pragma unroll
  for (int kb = 0; kb < 2; ++kb) {
    AFrag a0, a1;
    a0.h[0] = *(const i4*)(qba0 + kb * 32);
    a0.h[1] = *(const i4*)(qba0 + kb * 32 + 16);
    a1.h[0] = *(const i4*)(qba1 + kb * 32);
    a1.h[1] = *(const i4*)(qba1 + kb * 32 + 16);
#pragma unroll
    for (int u = 0; u < 4; ++u) {
      AFrag bu;
      bu.h[0] = *(const i4*)(bptr + kb * KT_BYTES + u * 1024);
      bu.h[1] = *(const i4*)(bptr + kb * KT_BYTES + u * 1024 + 16);
      acc[0][u] = wmma_bf16(a0, bu, acc[0][u]);
      acc[1][u] = wmma_bf16(a1, bu, acc[1][u]);
    }
  }

  // ---- writeback: C layout VGPR v -> M = v + (lane>=16 ? 8:0), N = lane&15 ----
#pragma unroll
  for (int t = 0; t < 2; ++t) {
#pragma unroll
    for (int u = 0; u < 4; ++u) {
      const int colg  = n0 + waveN * 64 + u * 16 + (lane & 15);
      const int rbase = m0 + waveM * 32 + t * 16 + ((lane & 16) ? 8 : 0);
#pragma unroll
      for (int v = 0; v < 8; ++v)
        out[(size_t)(rbase + v) * D_OUT + colg] = acc[t][u][v];
    }
  }
}

extern "C" void kernel_launch(void* const* d_in, const int* in_sizes, int n_in,
                              void* d_out, int out_size, void* d_ws, size_t ws_size,
                              hipStream_t stream) {
  (void)in_sizes; (void)n_in; (void)out_size; (void)ws_size;
  const float* x = (const float*)d_in[0];
  const float* q = (const float*)d_in[1];
  const float* W = (const float*)d_in[2];
  const float* b = (const float*)d_in[3];
  unsigned short* wp = (unsigned short*)d_ws;   // needs KTILES*16*32*32 B ≈ 8.4 MB
  float* out = (float*)d_out;

  const int packItems = KTILES * 16 * 32;
  pack_w<<<(packItems + 255) / 256, 256, 0, stream>>>(W, b, wp);

  dim3 grid(2, 64);   // 2 N-blocks of 128 cols x 64 M-blocks of 64 rows
  dynw_gemm<<<grid, 128, 0, stream>>>(x, q, wp, out);
}
